// GraphAttentionLayer_62835371540595
// MI455X (gfx1250) — matrix operand order repaired and checked by reading, not compiled, not run on hardware
//
#include <hip/hip_runtime.h>
#include <hip/hip_bf16.h>

// ---------------------------------------------------------------------------
// GATConv forward for MI455X (gfx1250, wave32).
//   h = x @ W           -> bf16 WMMA (v_wmma_f32_16x16x32_bf16), fp32 accum
//   a_s/a_d             -> per-(node,head) dot products
//   segment softmax     -> 3 edge passes (atomicMax key / exp+atomicAdd / scatter)
//   LayerNorm + ELU     -> one wave per node, shfl reduction (wave32)
// ---------------------------------------------------------------------------

typedef __attribute__((ext_vector_type(16))) __bf16 v16bf;
typedef __attribute__((ext_vector_type(8)))  __bf16 v8bf;
typedef __attribute__((ext_vector_type(8)))  float  v8f;

#define IN_CH    128
#define HEADS    4
#define OUT_CH   64
#define HC       256
#define NEG_SLOPE 0.2f
#define LN_EPS    1e-5f

// ---------------- helpers ----------------
__device__ __forceinline__ unsigned fkey(float f) {
    unsigned u = __float_as_uint(f);
    return (u & 0x80000000u) ? ~u : (u | 0x80000000u);
}
__device__ __forceinline__ float kinv(unsigned k) {
    unsigned u = (k & 0x80000000u) ? (k & 0x7fffffffu) : ~k;
    return __uint_as_float(u);
}

// ---------------- init / convert ----------------
__global__ void zero_bufs(float* __restrict__ out, float* __restrict__ ssum,
                          unsigned* __restrict__ mkey, size_t nout, int nm) {
    size_t i = (size_t)blockIdx.x * blockDim.x + threadIdx.x;
    size_t stride = (size_t)gridDim.x * blockDim.x;
    for (size_t j = i; j < nout; j += stride) out[j] = 0.0f;
    for (size_t j = i; j < (size_t)nm; j += stride) { ssum[j] = 0.0f; mkey[j] = 0u; }
}

__global__ void cvt_x_bf16(const float* __restrict__ x, __bf16* __restrict__ xb, size_t n) {
    size_t i = (size_t)blockIdx.x * blockDim.x + threadIdx.x;
    size_t stride = (size_t)gridDim.x * blockDim.x;
    for (size_t j = i; j < n; j += stride) xb[j] = (__bf16)x[j];
}

// W is [128][256] row-major fp32 -> Wt [256][128] bf16 (column of W contiguous in K)
__global__ void cvt_w_bf16_t(const float* __restrict__ W, __bf16* __restrict__ wt) {
    int i = blockIdx.x * blockDim.x + threadIdx.x;
    if (i >= IN_CH * HC) return;
    int k = i / HC, n = i % HC;
    wt[(size_t)n * IN_CH + k] = (__bf16)W[i];
}

// ---------------- GEMM: h = x @ W via WMMA bf16 ----------------
// one wave -> one 16x16 tile of h; K loop 128 in steps of 32.
__global__ __launch_bounds__(256) void gemm_wmma(const __bf16* __restrict__ xb,
                                                 const __bf16* __restrict__ wt,
                                                 float* __restrict__ hfeat, int nrows) {
    const int lane = threadIdx.x & 31;
    const int gw   = blockIdx.x * 8 + (threadIdx.x >> 5);
    const int rowTile = gw >> 4;          // HC/16 = 16 column tiles
    const int colTile = gw & 15;
    if (rowTile * 16 >= nrows) return;    // wave-uniform: EXEC stays all-ones
    const int half = lane >> 4;
    const int mr   = lane & 15;

    const __bf16* arow = xb + (size_t)(rowTile * 16 + mr) * IN_CH;   // A row
    const __bf16* brow = wt + (size_t)(colTile * 16 + mr) * IN_CH;   // B column

    v8f c = {};
#pragma unroll
    for (int k0 = 0; k0 < IN_CH; k0 += 32) {
        // A 16x32 bf16: lane(row m) holds contiguous K-half [16*half, 16*half+16)
        v16bf A = *(const v16bf*)(arow + k0 + half * 16);
        // B 32x16 bf16: lane(col n) holds contiguous K-half [16*half, 16*half+16)
        v16bf B = *(const v16bf*)(brow + k0 + half * 16);
        c = __builtin_amdgcn_wmma_f32_16x16x32_bf16(false, A, false, B,
                                                    (short)0, c, false, false);
    }
    // D layout: VGPR r -> row (r + 8*half), col (lane&15)
    float* outp = hfeat + (size_t)(rowTile * 16 + half * 8) * HC + colTile * 16 + (lane & 15);
#pragma unroll
    for (int r = 0; r < 8; ++r) outp[(size_t)r * HC] = c[r];
}

// ---------------- per-node attention halves ----------------
__global__ void att_dots(const float* __restrict__ hfeat,
                         const float* __restrict__ att_src,
                         const float* __restrict__ att_dst,
                         float* __restrict__ a_s, float* __restrict__ a_d, int n) {
    int t = blockIdx.x * blockDim.x + threadIdx.x;
    if (t >= n * HEADS) return;
    int node = t >> 2, head = t & 3;
    const float* hp = hfeat + (size_t)node * HC + head * OUT_CH;
    const float* as = att_src + head * OUT_CH;
    const float* ad = att_dst + head * OUT_CH;
    float s = 0.0f, d = 0.0f;
#pragma unroll 8
    for (int c = 0; c < OUT_CH; ++c) { float v = hp[c]; s += v * as[c]; d += v * ad[c]; }
    a_s[t] = s; a_d[t] = d;
}

// ---------------- edge pass 1: segment max (float atomicMax via uint key) ----
__global__ void edge_max(const int* __restrict__ ei,
                         const float* __restrict__ a_s, const float* __restrict__ a_d,
                         unsigned* __restrict__ mkey, int E_, int N_) {
    int idx = blockIdx.x * blockDim.x + threadIdx.x;
    if (idx >= E_ + N_) return;
    int src, dst;
    if (idx < E_) { src = ei[idx]; dst = ei[E_ + idx]; } else { src = dst = idx - E_; }
    const float4 as = *(const float4*)(a_s + (size_t)src * 4);
    const float4 ad = *(const float4*)(a_d + (size_t)dst * 4);
    float e[4] = {as.x + ad.x, as.y + ad.y, as.z + ad.z, as.w + ad.w};
#pragma unroll
    for (int h = 0; h < HEADS; ++h) {
        float v = e[h] > 0.0f ? e[h] : NEG_SLOPE * e[h];
        atomicMax(&mkey[(size_t)dst * 4 + h], fkey(v));
    }
}

// ---------------- edge pass 2: p = exp(e - m), s = segsum(p) ----------------
__global__ void edge_exp(const int* __restrict__ ei,
                         const float* __restrict__ a_s, const float* __restrict__ a_d,
                         const unsigned* __restrict__ mkey,
                         float* __restrict__ ssum, float* __restrict__ pbuf,
                         int E_, int N_) {
    int idx = blockIdx.x * blockDim.x + threadIdx.x;
    if (idx >= E_ + N_) return;
    int src, dst;
    if (idx < E_) { src = ei[idx]; dst = ei[E_ + idx]; } else { src = dst = idx - E_; }
    const float4 as = *(const float4*)(a_s + (size_t)src * 4);
    const float4 ad = *(const float4*)(a_d + (size_t)dst * 4);
    float e[4] = {as.x + ad.x, as.y + ad.y, as.z + ad.z, as.w + ad.w};
#pragma unroll
    for (int h = 0; h < HEADS; ++h) {
        float v = e[h] > 0.0f ? e[h] : NEG_SLOPE * e[h];
        float m = kinv(mkey[(size_t)dst * 4 + h]);
        float p = __expf(v - m);
        pbuf[(size_t)idx * 4 + h] = p;
        atomicAdd(&ssum[(size_t)dst * 4 + h], p);
    }
}

// ---------------- edge pass 3: weighted scatter-add (wave per edge) ----------
__global__ __launch_bounds__(256) void edge_scatter(const int* __restrict__ ei,
                                                    const float* __restrict__ hfeat,
                                                    const float* __restrict__ pbuf,
                                                    const float* __restrict__ ssum,
                                                    float* __restrict__ out,
                                                    int E_, int N_) {
    const int lane = threadIdx.x & 31;
    const int gw   = blockIdx.x * 8 + (threadIdx.x >> 5);
    if (gw >= E_ + N_) return;
    int src, dst;
    if (gw < E_) { src = ei[gw]; dst = ei[E_ + gw]; } else { src = dst = gw - E_; }
    const int head = lane >> 3;  // 8 lanes cover one 64-ch head
    float alpha = pbuf[(size_t)gw * 4 + head] / (ssum[(size_t)dst * 4 + head] + 1e-16f);
    const float4* hp = (const float4*)(hfeat + (size_t)src * HC + lane * 8);
    float4 v0 = hp[0], v1 = hp[1];
    float* op = out + (size_t)dst * HC + lane * 8;
    atomicAdd(op + 0, v0.x * alpha); atomicAdd(op + 1, v0.y * alpha);
    atomicAdd(op + 2, v0.z * alpha); atomicAdd(op + 3, v0.w * alpha);
    atomicAdd(op + 4, v1.x * alpha); atomicAdd(op + 5, v1.y * alpha);
    atomicAdd(op + 6, v1.z * alpha); atomicAdd(op + 7, v1.w * alpha);
}

// ---------------- LayerNorm + ELU (wave per node, wave32 shfl reduce) --------
__global__ __launch_bounds__(256) void ln_elu(float* __restrict__ out,
                                              const float* __restrict__ bias,
                                              const float* __restrict__ gamma,
                                              const float* __restrict__ beta, int n) {
    const int lane = threadIdx.x & 31;
    const int node = blockIdx.x * 8 + (threadIdx.x >> 5);
    if (node >= n) return;
    float* op = out + (size_t)node * HC + lane * 8;
    float4 v0 = *(float4*)op, v1 = *(float4*)(op + 4);
    const float* bp = bias + lane * 8;
    float4 b0 = *(const float4*)bp, b1 = *(const float4*)(bp + 4);
    float v[8] = {v0.x + b0.x, v0.y + b0.y, v0.z + b0.z, v0.w + b0.w,
                  v1.x + b1.x, v1.y + b1.y, v1.z + b1.z, v1.w + b1.w};
    float s1 = 0.0f, s2 = 0.0f;
#pragma unroll
    for (int j = 0; j < 8; ++j) { s1 += v[j]; s2 += v[j] * v[j]; }
#pragma unroll
    for (int off = 16; off > 0; off >>= 1) {
        s1 += __shfl_xor(s1, off, 32);
        s2 += __shfl_xor(s2, off, 32);
    }
    float mean = s1 * (1.0f / HC);
    float var  = s2 * (1.0f / HC) - mean * mean;
    float rstd = rsqrtf(var + LN_EPS);
    const float* gp = gamma + lane * 8;
    const float* tp = beta  + lane * 8;
    float r[8];
#pragma unroll
    for (int j = 0; j < 8; ++j) {
        float nv = (v[j] - mean) * rstd * gp[j] + tp[j];
        r[j] = nv > 0.0f ? nv : expm1f(nv);  // ELU(alpha=1)
    }
    *(float4*)op       = make_float4(r[0], r[1], r[2], r[3]);
    *(float4*)(op + 4) = make_float4(r[4], r[5], r[6], r[7]);
}

// ---------------------------------------------------------------------------
extern "C" void kernel_launch(void* const* d_in, const int* in_sizes, int n_in,
                              void* d_out, int out_size, void* d_ws, size_t ws_size,
                              hipStream_t stream) {
    const float* x        = (const float*)d_in[0];
    const int*   ei       = (const int*)  d_in[1];   // [2,E] int32
    const float* W        = (const float*)d_in[2];
    const float* att_src  = (const float*)d_in[3];
    const float* att_dst  = (const float*)d_in[4];
    const float* bias     = (const float*)d_in[5];
    const float* gamma    = (const float*)d_in[6];
    const float* beta     = (const float*)d_in[7];
    float* out = (float*)d_out;

    const int N = in_sizes[0] / IN_CH;
    const int E = in_sizes[1] / 2;
    const int T = E + N;                       // edges incl. self loops

    // carve workspace (256B aligned segments)
    char* w = (char*)d_ws;
    auto carve = [&](size_t bytes) -> char* {
        char* p = w; w += (bytes + 255) & ~(size_t)255; return p;
    };
    __bf16*   xb    = (__bf16*)  carve((size_t)N * IN_CH * 2);
    __bf16*   wt    = (__bf16*)  carve((size_t)HC * IN_CH * 2);
    float*    hfeat = (float*)   carve((size_t)N * HC * 4);
    float*    a_s   = (float*)   carve((size_t)N * HEADS * 4);
    float*    a_d   = (float*)   carve((size_t)N * HEADS * 4);
    unsigned* mkey  = (unsigned*)carve((size_t)N * HEADS * 4);
    float*    ssum  = (float*)   carve((size_t)N * HEADS * 4);
    float*    pbuf  = (float*)   carve((size_t)T * HEADS * 4);

    // 0) zero accumulation buffers (d_out is poisoned by harness)
    zero_bufs<<<4096, 256, 0, stream>>>(out, ssum, mkey, (size_t)N * HC, N * HEADS);

    // 1) fp32 -> bf16 conversions (W also transposed to [HC][IN_CH])
    cvt_x_bf16<<<4096, 256, 0, stream>>>(x, xb, (size_t)N * IN_CH);
    cvt_w_bf16_t<<<(IN_CH * HC + 255) / 256, 256, 0, stream>>>(W, wt);

    // 2) h = x @ W  (WMMA bf16, fp32 accum); one wave per 16x16 tile
    {
        int rowTiles = (N + 15) / 16;
        int waves = rowTiles * (HC / 16);
        gemm_wmma<<<(waves + 7) / 8, 256, 0, stream>>>(xb, wt, hfeat, N);
    }

    // 3) per-node attention halves
    att_dots<<<(N * HEADS + 255) / 256, 256, 0, stream>>>(hfeat, att_src, att_dst, a_s, a_d, N);

    // 4) segment softmax over incoming edges
    edge_max<<<(T + 255) / 256, 256, 0, stream>>>(ei, a_s, a_d, mkey, E, N);
    edge_exp<<<(T + 255) / 256, 256, 0, stream>>>(ei, a_s, a_d, mkey, ssum, pbuf, E, N);

    // 5) weighted scatter-add (wave per edge, 8 fp32 atomics/lane -> L2)
    edge_scatter<<<(T + 7) / 8, 256, 0, stream>>>(ei, hfeat, pbuf, ssum, out, E, N);

    // 6) LayerNorm + ELU in-place on d_out (wave per node)
    ln_elu<<<(N + 7) / 8, 256, 0, stream>>>(out, bias, gamma, beta, N);
}
